// Model_1236950581448
// MI455X (gfx1250) — compile-verified, matrix operand order checked by reading, not run on hardware
//
#include <hip/hip_runtime.h>
#include <hip/hip_bf16.h>
#include <math.h>

typedef _Float16 f16;
typedef _Float16 v16h __attribute__((ext_vector_type(16)));
typedef _Float16 v8h  __attribute__((ext_vector_type(8)));
typedef float    v8f  __attribute__((ext_vector_type(8)));

#define SEQ   2048
#define EMB   256
#define HH    256
#define G4    1024   /* 4*HH */
#define NTAGS 12
#define START_TAG 10
#define STOP_TAG  11
#define NEGV  (-10000.0f)

// LDS-resident half of Whh: 512 rows, padded row stride to kill bank conflicts.
// 264 f16 = 528 B = 132 dwords per row -> bank offset (4*row)%64, distinct for
// the 16 rows addressed by one WMMA A-fragment.
#define LROWS    512
#define LSTRIDE  264                       /* f16 elements per padded row   */
#define LDS_WHH_BYTES   (LROWS * LSTRIDE * 2)      /* 270336 B              */
#define LDS_H_OFF       LDS_WHH_BYTES              /* f16 h[256]  (512 B)   */
#define LDS_G_OFF       (LDS_WHH_BYTES + 512)      /* f32 g[1024] (4096 B)  */
#define LDS_TOTAL       (LDS_G_OFF + G4 * 4)       /* 274944 B < 320 KB     */

__device__ __forceinline__ v16h cat8(v8h lo, v8h hi) {
  return __builtin_shufflevector(lo, hi, 0,1,2,3,4,5,6,7,8,9,10,11,12,13,14,15);
}

__device__ __forceinline__ v8f wmma_f16(v16h a, v16h b, v8f c) {
  // v_wmma_f32_16x16x32_f16 : D = A(16x32 f16) * B(32x16 f16) + C(16x16 f32)
  return __builtin_amdgcn_wmma_f32_16x16x32_f16(false, a, false, b,
                                                (short)0, c, false, false);
}

// ---------------------------------------------------------------------------
// Kernel 1: gather embeddings to f16, cast Wih/Whh (both dirs) to f16.
// ---------------------------------------------------------------------------
__global__ void prep_kernel(const int* __restrict__ sentence,
                            const float* __restrict__ embed,
                            const float* __restrict__ WihF,
                            const float* __restrict__ WihB,
                            const float* __restrict__ WhhF,
                            const float* __restrict__ WhhB,
                            f16* __restrict__ xh,
                            f16* __restrict__ wihh,
                            f16* __restrict__ whhh) {
  int idx = blockIdx.x * blockDim.x + threadIdx.x;
  const int NX = SEQ * EMB;      // 524288
  const int NW = 2 * G4 * EMB;   // 524288
  if (idx < NX) {
    int t = idx / EMB, e = idx % EMB;
    xh[idx] = (f16)embed[(size_t)sentence[t] * EMB + e];
  } else if (idx < NX + NW) {
    int j = idx - NX;
    float v = (j < G4 * EMB) ? WihF[j] : WihB[j - G4 * EMB];
    wihh[j] = (f16)v;
  } else if (idx < NX + 2 * NW) {
    int j = idx - NX - NW;
    float v = (j < G4 * EMB) ? WhhF[j] : WhhB[j - G4 * EMB];
    whhh[j] = (f16)v;
  }
}

// ---------------------------------------------------------------------------
// Kernel 2: pre[dir][t][g] = x @ Wih^T + bih + bhh  via WMMA.
// One wave per 16x16 output tile; K = 256 in 8 chunks of 32.
// A-layout per ISA (16-bit 16x32): lane = M | (K-half<<4); two 16B chunks.
// B = Wih^T  =>  B column n == Wih row (g0+n), identical load pattern.
// ---------------------------------------------------------------------------
__global__ void input_proj_kernel(const f16* __restrict__ xh,
                                  const f16* __restrict__ wihh,
                                  const float* __restrict__ bihF,
                                  const float* __restrict__ bhhF,
                                  const float* __restrict__ bihB,
                                  const float* __restrict__ bhhB,
                                  float* __restrict__ pre) {
  int wid  = blockIdx.x * (blockDim.x >> 5) + (threadIdx.x >> 5);
  int lane = threadIdx.x & 31;
  int dir   = wid >> 13;         // 2 * 128 * 64 waves total
  int ttile = (wid >> 6) & 127;
  int gtile = wid & 63;
  int t0 = ttile * 16, g0 = gtile * 16;

  int row   = lane & 15;
  int khalf = (lane >= 16) ? 8 : 0;

  const f16* A0 = xh + (size_t)(t0 + row) * EMB;
  const f16* B0 = wihh + (size_t)dir * G4 * EMB + (size_t)(g0 + row) * EMB;

  v8f acc = {};
#pragma unroll
  for (int kc = 0; kc < 8; ++kc) {
    int k0 = kc * 32 + khalf;
    v8h a_lo = *(const v8h*)(A0 + k0);
    v8h a_hi = *(const v8h*)(A0 + k0 + 16);
    v8h b_lo = *(const v8h*)(B0 + k0);
    v8h b_hi = *(const v8h*)(B0 + k0 + 16);
    acc = wmma_f16(cat8(a_lo, a_hi), cat8(b_lo, b_hi), acc);
  }

  int col = lane & 15;
  int g   = g0 + col;
  float bias = dir ? (bihB[g] + bhhB[g]) : (bihF[g] + bhhF[g]);
  int rowoff = (lane < 16) ? 0 : 8;
  float* out = pre + (size_t)dir * SEQ * G4 + (size_t)(t0 + rowoff) * G4 + g;
#pragma unroll
  for (int r = 0; r < 8; ++r)
    out[(size_t)r * G4] = acc[r] + bias;
}

// ---------------------------------------------------------------------------
// Kernel 3: sequential LSTM recurrence, one block (= one WGP) per direction.
// 32 waves; wave w owns gate rows [32w, 32w+32).
//  - Gate rows 0..511 of Whh live in LDS (bank-conflict-padded), loaded once.
//  - Gate rows 512..1023 stream from the 192MB L2 each step.
//  - h is kept in LDS as f16 and broadcast into all 16 B-columns, so every
//    column of D equals Whh@h; lanes 0/16 scatter the result to LDS.
// ---------------------------------------------------------------------------
__global__ void __launch_bounds__(1024)
lstm_kernel(const f16* __restrict__ whhh,
            const float* __restrict__ pre,
            const float* __restrict__ h0,
            const float* __restrict__ c0,
            float* __restrict__ hcat) {
  extern __shared__ __align__(16) char smem[];
  f16*   wlds = (f16*)smem;                  // padded [512][264]
  f16*   hsh  = (f16*)(smem + LDS_H_OFF);    // h[256]
  float* gsh  = (float*)(smem + LDS_G_OFF);  // gates[1024]

  int dir   = blockIdx.x;
  int tid   = threadIdx.x;
  int wavei = tid >> 5;
  int lane  = tid & 31;

  const f16*   W    = whhh + (size_t)dir * G4 * HH;
  const float* preD = pre + (size_t)dir * SEQ * G4;

  // ---- stage Whh rows 0..511 into LDS (16B vectors, padded rows) ----
  for (int ii = tid; ii < LROWS * (HH / 8); ii += 1024) {
    int r = ii >> 5;            // /32 chunks per row
    int c = ii & 31;
    *(v8h*)(wlds + (size_t)r * LSTRIDE + c * 8) =
        *(const v8h*)(W + (size_t)r * HH + c * 8);
  }
  if (tid < HH) hsh[tid] = (f16)h0[dir * HH + tid];
  float c = (tid < HH) ? c0[dir * HH + tid] : 0.0f;
  __syncthreads();

  int row   = lane & 15;
  int khalf = (lane >= 16) ? 8 : 0;

  for (int step = 0; step < SEQ; ++step) {
    int t = dir ? (SEQ - 1 - step) : step;

    // --- matvec g = Whh @ h on the matrix pipe (wave-uniform exec) ---
#pragma unroll
    for (int mt = 0; mt < 2; ++mt) {
      int rowbase = wavei * 32 + mt * 16;
      v8f acc = {};
      if (wavei < 16) {                       // LDS-resident half
        const f16* Wr = wlds + (size_t)(rowbase + row) * LSTRIDE;
#pragma unroll
        for (int kc = 0; kc < 8; ++kc) {
          int k0 = kc * 32 + khalf;
          v8h a_lo = *(const v8h*)(Wr + k0);          // ds_load_b128
          v8h a_hi = *(const v8h*)(Wr + k0 + 16);
          v8h b_lo = *(const v8h*)(hsh + k0);         // LDS broadcast
          v8h b_hi = *(const v8h*)(hsh + k0 + 16);
          acc = wmma_f16(cat8(a_lo, a_hi), cat8(b_lo, b_hi), acc);
        }
      } else {                                // L2-streamed half
        const f16* Wr = W + (size_t)(rowbase + row) * HH;
#pragma unroll
        for (int kc = 0; kc < 8; ++kc) {
          int k0 = kc * 32 + khalf;
          v8h a_lo = *(const v8h*)(Wr + k0);          // global_load_b128
          v8h a_hi = *(const v8h*)(Wr + k0 + 16);
          v8h b_lo = *(const v8h*)(hsh + k0);
          v8h b_hi = *(const v8h*)(hsh + k0 + 16);
          acc = wmma_f16(cat8(a_lo, a_hi), cat8(b_lo, b_hi), acc);
        }
      }
      // every column of D equals the matvec; lanes 0/16 hold rows 0..7/8..15
      if (lane == 0) {
#pragma unroll
        for (int r = 0; r < 8; ++r) gsh[rowbase + r] = acc[r];
      } else if (lane == 16) {
#pragma unroll
        for (int r = 0; r < 8; ++r) gsh[rowbase + 8 + r] = acc[r];
      }
    }
    __syncthreads();

    // --- pointwise gate math (threads 0..255), PyTorch order i,f,g,o ---
    if (tid < HH) {
      const float* pt = preD + (size_t)t * G4;
      float gi = gsh[tid]       + pt[tid];
      float gf = gsh[tid + 256] + pt[tid + 256];
      float gg = gsh[tid + 512] + pt[tid + 512];
      float go = gsh[tid + 768] + pt[tid + 768];
      float i = 1.0f / (1.0f + expf(-gi));
      float f = 1.0f / (1.0f + expf(-gf));
      float g = tanhf(gg);
      float o = 1.0f / (1.0f + expf(-go));
      c = f * c + i * g;
      float hn = o * tanhf(c);
      hcat[(size_t)t * (2 * HH) + dir * HH + tid] = hn;
      hsh[tid] = (f16)hn;
    }
    __syncthreads();
  }
}

// ---------------------------------------------------------------------------
// Kernel 4: feats[t][k] = dot(hcat[t], W_out[k]) + b_out[k]
// ---------------------------------------------------------------------------
__global__ void outproj_kernel(const float* __restrict__ hcat,
                               const float* __restrict__ Wout,
                               const float* __restrict__ bout,
                               float* __restrict__ feats) {
  int idx = blockIdx.x * blockDim.x + threadIdx.x;
  if (idx >= SEQ * NTAGS) return;
  int t = idx / NTAGS, k = idx % NTAGS;
  const float* h = hcat + (size_t)t * (2 * HH);
  const float* w = Wout + (size_t)k * (2 * HH);
  float s = bout[k];
  for (int j = 0; j < 2 * HH; ++j) s = fmaf(h[j], w[j], s);
  feats[idx] = s;
}

// ---------------------------------------------------------------------------
// Kernel 5: CRF forward scan + gold path score; single wave32.
// ---------------------------------------------------------------------------
__global__ void crf_kernel(const float* __restrict__ feats,
                           const float* __restrict__ trans,
                           const int* __restrict__ tags,
                           float* __restrict__ out) {
  int lane = threadIdx.x;               // 32 lanes
  __shared__ float fv[NTAGS];
  __shared__ float goldsh;
  int i = (lane < NTAGS) ? lane : (NTAGS - 1);

  float tr[NTAGS];
#pragma unroll
  for (int j = 0; j < NTAGS; ++j) tr[j] = trans[i * NTAGS + j];

  // gold path score, wave-parallel over t
  float gp = 0.0f;
  for (int t = lane; t < SEQ; t += 32) {
    int tag  = tags[t];
    int prev = (t == 0) ? START_TAG : tags[t - 1];
    gp += trans[tag * NTAGS + prev] + feats[t * NTAGS + tag];
  }
#pragma unroll
  for (int off = 16; off > 0; off >>= 1) gp += __shfl_down(gp, off, 32);
  if (lane == 0) goldsh = gp + trans[STOP_TAG * NTAGS + tags[SEQ - 1]];

  if (lane < NTAGS) fv[lane] = (lane == START_TAG) ? 0.0f : NEGV;
  __syncthreads();

  for (int t = 0; t < SEQ; ++t) {
    float vj[NTAGS];
    float m = -1e30f;
#pragma unroll
    for (int j = 0; j < NTAGS; ++j) {
      vj[j] = fv[j] + tr[j];
      m = fmaxf(m, vj[j]);
    }
    float s = 0.0f;
#pragma unroll
    for (int j = 0; j < NTAGS; ++j) s += expf(vj[j] - m);
    float nf = m + logf(s) + feats[t * NTAGS + i];
    __syncthreads();
    if (lane < NTAGS) fv[lane] = nf;
    __syncthreads();
  }

  if (lane == 0) {
    float m = -1e30f;
    float vj[NTAGS];
#pragma unroll
    for (int j = 0; j < NTAGS; ++j) {
      vj[j] = fv[j] + trans[STOP_TAG * NTAGS + j];
      m = fmaxf(m, vj[j]);
    }
    float s = 0.0f;
#pragma unroll
    for (int j = 0; j < NTAGS; ++j) s += expf(vj[j] - m);
    out[0] = (m + logf(s)) - goldsh;
  }
}

// ---------------------------------------------------------------------------
extern "C" void kernel_launch(void* const* d_in, const int* in_sizes, int n_in,
                              void* d_out, int out_size, void* d_ws, size_t ws_size,
                              hipStream_t stream) {
  (void)in_sizes; (void)n_in; (void)out_size; (void)ws_size;
  const int*   sentence = (const int*)d_in[0];
  const int*   tags     = (const int*)d_in[1];
  const float* embed    = (const float*)d_in[2];
  const float* WihF = (const float*)d_in[3];
  const float* WhhF = (const float*)d_in[4];
  const float* bihF = (const float*)d_in[5];
  const float* bhhF = (const float*)d_in[6];
  const float* WihB = (const float*)d_in[7];
  const float* WhhB = (const float*)d_in[8];
  const float* bihB = (const float*)d_in[9];
  const float* bhhB = (const float*)d_in[10];
  const float* Wout = (const float*)d_in[11];
  const float* bout = (const float*)d_in[12];
  const float* h0   = (const float*)d_in[13];
  const float* c0   = (const float*)d_in[14];
  const float* trans= (const float*)d_in[15];
  float* out = (float*)d_out;

  char* ws = (char*)d_ws;
  size_t off = 0;
  auto alloc = [&](size_t bytes) -> void* {
    void* p = ws + off;
    off = (off + bytes + 255) & ~(size_t)255;
    return p;
  };
  f16*   xh    = (f16*)alloc((size_t)SEQ * EMB * sizeof(f16));        // 1 MB
  f16*   wihh  = (f16*)alloc((size_t)2 * G4 * EMB * sizeof(f16));     // 1 MB
  f16*   whhh  = (f16*)alloc((size_t)2 * G4 * EMB * sizeof(f16));     // 1 MB
  float* pre   = (float*)alloc((size_t)2 * SEQ * G4 * sizeof(float)); // 16 MB
  float* hcat  = (float*)alloc((size_t)SEQ * 2 * HH * sizeof(float)); // 4 MB
  float* feats = (float*)alloc((size_t)SEQ * NTAGS * sizeof(float));  // 96 KB

  // 1) gather + f16 casts: 3 * 524288 elements
  const int prep_total = 3 * SEQ * EMB;
  prep_kernel<<<(prep_total + 255) / 256, 256, 0, stream>>>(
      sentence, embed, WihF, WihB, WhhF, WhhB, xh, wihh, whhh);
  // 2) input projection GEMM: 2*128*64 = 16384 waves, 8 waves/block
  input_proj_kernel<<<2048, 256, 0, stream>>>(xh, wihh, bihF, bhhF, bihB, bhhB, pre);
  // 3) recurrence: one persistent block per direction; 275 KB dynamic LDS
  lstm_kernel<<<2, 1024, LDS_TOTAL, stream>>>(whhh, pre, h0, c0, hcat);
  // 4) output projection
  outproj_kernel<<<(SEQ * NTAGS + 255) / 256, 256, 0, stream>>>(hcat, Wout, bout, feats);
  // 5) CRF scan + gold score -> scalar loss
  crf_kernel<<<1, 32, 0, stream>>>(feats, trans, tags, out);
}
